// LeechSelfAttention_70282844831783
// MI455X (gfx1250) — matrix operand order repaired and checked by reading, not compiled, not run on hardware
//
#include <hip/hip_runtime.h>
#include <hip/hip_bf16.h>
#include <math.h>

// ---------------------------------------------------------------------------
// LeechSelfAttention for MI455X (gfx1250, wave32, WMMA, async LDS copies).
// Pipeline:
//   1) f32->bf16 convert of x and w_out
//   2) fold 24x24 leech rotation into w_qkv rows (q,k sections), emit bf16
//   3) QKV GEMM (bf16 WMMA, f32 acc, double-buffered async-LDS tiles)
//      -> scatter q (pre-scaled), k, v as [B,H,S,96]
//   4) flash attention per (b,h): async K tile, transposed V tile,
//      WMMA QK^T, online softmax, WMMA P*V
//   5) out GEMM (bf16 WMMA) -> f32 d_out
// ---------------------------------------------------------------------------

typedef __bf16 bf16_t;
typedef __attribute__((ext_vector_type(16))) __bf16 bf16x16;
typedef __attribute__((ext_vector_type(8)))  float  f32x8;

#define D_MODEL  1536
#define N_HEADS  16
#define HEAD_DIM 96
#define SEQ      2048
#define BATCH    2
#define TOKENS   (BATCH * SEQ)      // 4096
#define QKV_OUT  (3 * D_MODEL)      // 4608
#define QSCALE   0.10206207f        // 96^-0.5

// ---------------------- async global->LDS copy (CDNA5) ---------------------
// GLOBAL_LOAD_ASYNC_TO_LDS_B128: per-lane 16-byte global->LDS copy, tracked by ASYNCcnt.
__device__ __forceinline__ void async_b128(const bf16_t* g, bf16_t* l) {
  unsigned lds_off = (unsigned)(uintptr_t)l;              // AS3 offset = low 32b
  unsigned long long ga = (unsigned long long)(uintptr_t)g;
  asm volatile("global_load_async_to_lds_b128 %0, %1, off"
               :: "v"(lds_off), "v"(ga) : "memory");
}
__device__ __forceinline__ void wait_async_le4() {
  asm volatile("s_wait_asynccnt 4" ::: "memory");
}
__device__ __forceinline__ void wait_async_0() {
  asm volatile("s_wait_asynccnt 0" ::: "memory");
}

// ------------------------- WMMA fragment loaders ---------------------------
// A (16x32 bf16): lane row = lane&15; lane-half selects K sub-blocks
__device__ __forceinline__ bf16x16 frag_a_ld(const bf16_t* p, int ld, int koff) {
  const int lane = threadIdx.x & 31;
  const int row  = lane & 15;
  const int kh   = lane >> 4;
  const bf16_t* r0 = p + (size_t)row * ld + koff + kh * 8;
  bf16x16 a;
#pragma unroll
  for (int i = 0; i < 8; ++i) a[i] = r0[i];
#pragma unroll
  for (int i = 0; i < 8; ++i) a[8 + i] = r0[16 + i];
  return a;
}

// B (32x16 bf16) from N-major storage (source row n holds the K axis): two
// 8-element K runs (same K order as frag_a_ld) -> 2x ds_load_b128 per fragment.
__device__ __forceinline__ bf16x16 frag_b_rowmajor(const bf16_t* p, int ld, int nbase, int koff) {
  const int lane = threadIdx.x & 31;
  const int n    = nbase + (lane & 15);
  const int kh   = lane >> 4;
  const bf16_t* r0 = p + (size_t)n * ld + koff + kh * 8;
  bf16x16 b;
#pragma unroll
  for (int i = 0; i < 8; ++i) b[i] = r0[i];
#pragma unroll
  for (int i = 0; i < 8; ++i) b[8 + i] = r0[16 + i];
  return b;
}

#define WMMA_BF16(a, b, c) \
  __builtin_amdgcn_wmma_f32_16x16x32_bf16(false, (a), false, (b), (short)0, (c), false, false)

// ----------------------------- prep kernels --------------------------------
__global__ void f32_to_bf16_kernel(const float* __restrict__ in, bf16_t* __restrict__ out, int n) {
  int i = blockIdx.x * blockDim.x + threadIdx.x;
  if (i < n) out[i] = (bf16_t)in[i];
}

// Fold blockwise 24x24 rotation into w_qkv rows. Rows [0,3072) (q,k) rotate,
// rows [3072,4608) (v) pass through. grid = (cols/128, 4608/24), block = 128.
__global__ void fold_rotation_kernel(const float* __restrict__ w_qkv,
                                     const float* __restrict__ leech,
                                     bf16_t* __restrict__ wrot) {
  __shared__ float Ls[24 * 24];
  const int tid = threadIdx.x;
  for (int i = tid; i < 576; i += 128) Ls[i] = leech[i];
  __syncthreads();

  const int col = blockIdx.x * 128 + tid;
  const int g   = blockIdx.y;           // 24-row group
  const size_t base = (size_t)g * 24 * D_MODEL + col;
  if (g < 128) {  // q,k sections: rotate
    float in[24];
#pragma unroll
    for (int j = 0; j < 24; ++j) in[j] = w_qkv[base + (size_t)j * D_MODEL];
#pragma unroll
    for (int jp = 0; jp < 24; ++jp) {
      float s = 0.f;
#pragma unroll
      for (int j = 0; j < 24; ++j) s += in[j] * Ls[j * 24 + jp];
      wrot[base + (size_t)jp * D_MODEL] = (bf16_t)s;
    }
  } else {        // v section: copy
#pragma unroll
    for (int j = 0; j < 24; ++j)
      wrot[base + (size_t)j * D_MODEL] = (bf16_t)w_qkv[base + (size_t)j * D_MODEL];
  }
}

// ------------------------------- GEMM --------------------------------------
// C[t,o] = sum_k A[t,k] * W[o,k].  M=4096 fixed, K=1536 fixed, N = NTOTAL.
// Block tile 128x128, 8 waves as 2(M)x4(N), each wave 64x32 = 4x2 WMMA frags.
// Both LDS tiles are row-major-along-K [row][k] with ld=40 (16B-aligned rows),
// filled by async global->LDS b128 copies and double-buffered.
template <int NTOTAL, bool QKV_EPI>
__global__ __launch_bounds__(256) void gemm_bf16_kernel(
    const bf16_t* __restrict__ A, const bf16_t* __restrict__ W,
    bf16_t* __restrict__ qb, bf16_t* __restrict__ kb, bf16_t* __restrict__ vb,
    float* __restrict__ outp) {
  __shared__ bf16_t As[2][128 * 40];
  __shared__ bf16_t Bs[2][128 * 40];

  const int tid   = threadIdx.x;
  const int mbase = blockIdx.y * 128;
  const int nbase = blockIdx.x * 128;
  const int w     = tid >> 5;
  const int wm    = (w >> 2) * 64;
  const int wn    = (w & 3) * 32;

  f32x8 acc[4][2];
#pragma unroll
  for (int fm = 0; fm < 4; ++fm)
#pragma unroll
    for (int fn = 0; fn < 2; ++fn)
      acc[fm][fn] = (f32x8){0.f, 0.f, 0.f, 0.f, 0.f, 0.f, 0.f, 0.f};

  // issue one tile's async loads: 2 b128 for A + 2 for B per thread
  auto issue_tile = [&](int buf, int k0) {
#pragma unroll
    for (int p = 0; p < 2; ++p) {
      int v = tid + p * 256;
      int row = v >> 2, kv = v & 3;
      async_b128(A + (size_t)(mbase + row) * D_MODEL + k0 + kv * 8,
                 &As[buf][row * 40 + kv * 8]);
      async_b128(W + (size_t)(nbase + row) * D_MODEL + k0 + kv * 8,
                 &Bs[buf][row * 40 + kv * 8]);
    }
  };

  issue_tile(0, 0);
  int cur = 0;
  for (int k0 = 0; k0 < D_MODEL; k0 += 32) {
    if (k0 + 32 < D_MODEL) {
      issue_tile(cur ^ 1, k0 + 32);  // prefetch next tile into other buffer
      wait_async_le4();              // in-order ASYNCcnt => current tile done
    } else {
      wait_async_0();
    }
    __syncthreads();

    bf16x16 af[4], bg[2];
#pragma unroll
    for (int fm = 0; fm < 4; ++fm)
      af[fm] = frag_a_ld(&As[cur][(wm + fm * 16) * 40], 40, 0);
#pragma unroll
    for (int fn = 0; fn < 2; ++fn)
      bg[fn] = frag_b_rowmajor(&Bs[cur][0], 40, wn + fn * 16, 0);
#pragma unroll
    for (int fm = 0; fm < 4; ++fm)
#pragma unroll
      for (int fn = 0; fn < 2; ++fn)
        acc[fm][fn] = WMMA_BF16(af[fm], bg[fn], acc[fm][fn]);
    __syncthreads();   // all waves done with buffer before it is re-filled
    cur ^= 1;
  }

  // epilogue: C layout row = r + 8*(lane>>4), col = nbase..+(lane&15)
  const int lane = tid & 31;
  const int hh = lane >> 4, nl = lane & 15;
#pragma unroll
  for (int fm = 0; fm < 4; ++fm)
#pragma unroll
    for (int fn = 0; fn < 2; ++fn)
#pragma unroll
      for (int r = 0; r < 8; ++r) {
        int t = mbase + wm + fm * 16 + r + 8 * hh;
        int o = nbase + wn + fn * 16 + nl;
        float val = acc[fm][fn][r];
        if constexpr (QKV_EPI) {
          int which = o / D_MODEL;
          int rem   = o - which * D_MODEL;
          int head  = rem / HEAD_DIM;
          int d     = rem - head * HEAD_DIM;
          int b = t >> 11, s = t & (SEQ - 1);
          size_t idx = (((size_t)b * N_HEADS + head) * SEQ + s) * HEAD_DIM + d;
          if (which == 0)      qb[idx] = (bf16_t)(val * QSCALE);
          else if (which == 1) kb[idx] = (bf16_t)val;
          else                 vb[idx] = (bf16_t)val;
        } else {
          outp[(size_t)t * NTOTAL + o] = val;
        }
      }
}

// --------------------------- flash attention -------------------------------
#define TQ 128
#define TK 64
#define KV_LD 104   // 96 + 8 pad (16B-aligned rows)
#define VT_LD 72    // 64 + 8 pad
#define P_LD  72    // 64 + 8 pad

__global__ __launch_bounds__(256) void flash_attn_kernel(
    const bf16_t* __restrict__ qb, const bf16_t* __restrict__ kb,
    const bf16_t* __restrict__ vb, bf16_t* __restrict__ attnb) {
  __shared__ bf16_t Ks[TK * KV_LD];        // [kv][d]  (async-filled)
  __shared__ bf16_t Vt[HEAD_DIM * VT_LD];  // [d][kv]  (transposed at load)
  __shared__ bf16_t Ps[8 * 16 * P_LD];     // per-wave P strips

  const int tid  = threadIdx.x;
  const int lane = tid & 31;
  const int w    = tid >> 5;          // 8 waves, each owns 16 q rows
  const int bh   = blockIdx.y;        // 0..31
  const int b    = bh >> 4, h = bh & 15;
  const int q0   = blockIdx.x * TQ + w * 16;

  const bf16_t* Q  = qb + ((size_t)bh * SEQ + q0) * HEAD_DIM;
  const bf16_t* Kg = kb + (size_t)bh * SEQ * HEAD_DIM;
  const bf16_t* Vg = vb + (size_t)bh * SEQ * HEAD_DIM;

  // Q fragments live in registers: 3 x (16x32) covering head_dim 96
  bf16x16 aq[3];
#pragma unroll
  for (int c = 0; c < 3; ++c) aq[c] = frag_a_ld(Q, HEAD_DIM, c * 32);

  f32x8 O[6];
#pragma unroll
  for (int f = 0; f < 6; ++f) O[f] = (f32x8){0.f, 0.f, 0.f, 0.f, 0.f, 0.f, 0.f, 0.f};
  float mrow[8], lrow[8];
#pragma unroll
  for (int r = 0; r < 8; ++r) { mrow[r] = -3.0e38f; lrow[r] = 0.f; }

  bf16_t* Pw = Ps + w * 16 * P_LD;

  for (int kv0 = 0; kv0 < SEQ; kv0 += TK) {
    // K tile: async global->LDS b128, [kv][d]; overlaps the V transpose below
#pragma unroll
    for (int p = 0; p < 3; ++p) {
      int v = tid + p * 256;
      int row = v / 12, c8 = v - row * 12;
      async_b128(Kg + (size_t)(kv0 + row) * HEAD_DIM + c8 * 8,
                 &Ks[row * KV_LD + c8 * 8]);
    }
    // V tile: sync load + transpose into [d][kv] so P*V reads are contiguous
#pragma unroll
    for (int p = 0; p < 3; ++p) {
      int v = tid + p * 256;
      int row = v / 12, c8 = v - row * 12;
      uint4 u = *reinterpret_cast<const uint4*>(
          Vg + (size_t)(kv0 + row) * HEAD_DIM + c8 * 8);
      bf16_t tmp[8];
      *reinterpret_cast<uint4*>(tmp) = u;
#pragma unroll
      for (int j = 0; j < 8; ++j) Vt[(c8 * 8 + j) * VT_LD + row] = tmp[j];
    }
    wait_async_0();
    __syncthreads();

    if (kv0 + TK < SEQ) {  // L2 warm-up for next tile (global_prefetch_b8)
      __builtin_prefetch(Kg + (size_t)(kv0 + TK + (tid & 63)) * HEAD_DIM, 0, 1);
      __builtin_prefetch(Vg + (size_t)(kv0 + TK + (tid & 63)) * HEAD_DIM, 0, 1);
    }

    // S = Q K^T (scale pre-folded into Q): 16 x 64, contraction over d=96
    f32x8 Sf[4];
#pragma unroll
    for (int nf = 0; nf < 4; ++nf) Sf[nf] = (f32x8){0.f, 0.f, 0.f, 0.f, 0.f, 0.f, 0.f, 0.f};
#pragma unroll
    for (int c = 0; c < 3; ++c)
#pragma unroll
      for (int nf = 0; nf < 4; ++nf) {
        bf16x16 bk = frag_b_rowmajor(Ks, KV_LD, nf * 16, c * 32);
        Sf[nf] = WMMA_BF16(aq[c], bk, Sf[nf]);
      }

    // online softmax: each half-wave's 16 lanes hold one row's 16 columns
    {
      const int hh = lane >> 4, nl = lane & 15;
      float alpha[8];
#pragma unroll
      for (int r = 0; r < 8; ++r) {
        float v = fmaxf(fmaxf(Sf[0][r], Sf[1][r]), fmaxf(Sf[2][r], Sf[3][r]));
#pragma unroll
        for (int msk = 1; msk <= 8; msk <<= 1) v = fmaxf(v, __shfl_xor(v, msk, 32));
        float mnew = fmaxf(mrow[r], v);
        alpha[r] = __expf(mrow[r] - mnew);
        mrow[r] = mnew;
      }
#pragma unroll
      for (int r = 0; r < 8; ++r) {
        float sum = 0.f;
#pragma unroll
        for (int nf = 0; nf < 4; ++nf) {
          float pv = __expf(Sf[nf][r] - mrow[r]);
          sum += pv;
          Pw[(r + 8 * hh) * P_LD + nf * 16 + nl] = (bf16_t)pv;  // C-layout -> LDS
        }
#pragma unroll
        for (int msk = 1; msk <= 8; msk <<= 1) sum += __shfl_xor(sum, msk, 32);
        lrow[r] = lrow[r] * alpha[r] + sum;
      }
#pragma unroll
      for (int f = 0; f < 6; ++f)
#pragma unroll
        for (int r = 0; r < 8; ++r) O[f][r] *= alpha[r];
    }

    // P strip is wave-private LDS: drain DS writes, then re-read as A frags
    asm volatile("s_wait_dscnt 0" ::: "memory");

    // O += P * V : contraction over kv = 64 (2 x K32), N = 96 (6 frags)
#pragma unroll
    for (int ks = 0; ks < 2; ++ks) {
      bf16x16 ap = frag_a_ld(Pw, P_LD, ks * 32);
#pragma unroll
      for (int f = 0; f < 6; ++f) {
        bf16x16 bv = frag_b_rowmajor(Vt, VT_LD, f * 16, ks * 32);
        O[f] = WMMA_BF16(ap, bv, O[f]);
      }
    }
    __syncthreads();
  }

  // normalize and store [b, s, h*96 + d] as bf16
  const int hh = lane >> 4, nl = lane & 15;
  float inv[8];
#pragma unroll
  for (int r = 0; r < 8; ++r) inv[r] = 1.f / lrow[r];
#pragma unroll
  for (int f = 0; f < 6; ++f)
#pragma unroll
    for (int r = 0; r < 8; ++r) {
      int s = q0 + r + 8 * hh;
      int d = f * 16 + nl;
      attnb[((size_t)b * SEQ + s) * D_MODEL + h * HEAD_DIM + d] =
          (bf16_t)(O[f][r] * inv[r]);
    }
}

// ------------------------------- launcher ----------------------------------
extern "C" void kernel_launch(void* const* d_in, const int* in_sizes, int n_in,
                              void* d_out, int out_size, void* d_ws, size_t ws_size,
                              hipStream_t stream) {
  (void)in_sizes; (void)n_in; (void)out_size; (void)ws_size;
  const float* x      = (const float*)d_in[0];
  const float* w_qkv  = (const float*)d_in[1];
  const float* w_out  = (const float*)d_in[2];
  const float* leech  = (const float*)d_in[3];
  float* out = (float*)d_out;

  char* ws = (char*)d_ws;
  size_t off = 0;
  auto take = [&](size_t elems) {
    bf16_t* p = (bf16_t*)(ws + off);
    off += ((elems * sizeof(bf16_t) + 255) & ~(size_t)255);
    return p;
  };
  bf16_t* xb    = take((size_t)TOKENS * D_MODEL);
  bf16_t* wqkvb = take((size_t)QKV_OUT * D_MODEL);
  bf16_t* woutb = take((size_t)D_MODEL * D_MODEL);
  bf16_t* qbuf  = take((size_t)TOKENS * D_MODEL);
  bf16_t* kbuf  = take((size_t)TOKENS * D_MODEL);
  bf16_t* vbuf  = take((size_t)TOKENS * D_MODEL);
  bf16_t* attnb = take((size_t)TOKENS * D_MODEL);

  const int n_x = TOKENS * D_MODEL;
  const int n_w = D_MODEL * D_MODEL;
  f32_to_bf16_kernel<<<(n_x + 255) / 256, 256, 0, stream>>>(x, xb, n_x);
  f32_to_bf16_kernel<<<(n_w + 255) / 256, 256, 0, stream>>>(w_out, woutb, n_w);
  fold_rotation_kernel<<<dim3(D_MODEL / 128, QKV_OUT / 24), 128, 0, stream>>>(
      w_qkv, leech, wqkvb);

  gemm_bf16_kernel<QKV_OUT, true><<<dim3(QKV_OUT / 128, TOKENS / 128), 256, 0, stream>>>(
      xb, wqkvb, qbuf, kbuf, vbuf, nullptr);

  flash_attn_kernel<<<dim3(SEQ / TQ, BATCH * N_HEADS), 256, 0, stream>>>(
      qbuf, kbuf, vbuf, attnb);

  gemm_bf16_kernel<D_MODEL, false><<<dim3(D_MODEL / 128, TOKENS / 128), 256, 0, stream>>>(
      attnb, woutb, nullptr, nullptr, nullptr, out);
}
